// GPTQLinear_4466765988095
// MI455X (gfx1250) — matrix operand order repaired and checked by reading, not compiled, not run on hardware
//
#include <hip/hip_runtime.h>

typedef __attribute__((ext_vector_type(16))) _Float16 v16h;
typedef __attribute__((ext_vector_type(8)))  _Float16 v8h;
typedef __attribute__((ext_vector_type(4)))  _Float16 v4h;
typedef __attribute__((ext_vector_type(2)))  _Float16 v2h;
typedef __attribute__((ext_vector_type(8)))  float    v8f;
typedef __attribute__((ext_vector_type(4)))  float    v4f;
typedef __attribute__((ext_vector_type(4)))  int      v4i;

#define TILE_M   128
#define TILE_N   128
#define TILE_K   32
#define LDS_STR  40   // halves per row: 32 data + 8 pad -> 20-dword stride, conflict-free b128

#if __has_builtin(__builtin_amdgcn_global_load_async_to_lds_b128) && \
    __has_builtin(__builtin_amdgcn_s_wait_asynccnt)
#define USE_ASYNC_LDS 1
// The builtin takes (int4 addrspace(1)*, int4 addrspace(3)*, imm offset, imm cpol)
typedef __attribute__((address_space(1))) void* as1_void_ptr;
typedef __attribute__((address_space(3))) void* as3_void_ptr;
typedef __attribute__((address_space(1))) v4i*  as1_v4i_ptr;
typedef __attribute__((address_space(3))) v4i*  as3_v4i_ptr;
#else
#define USE_ASYNC_LDS 0
#endif

// ---------------------------------------------------------------------------
// Pre-pass 1: x f32 -> f16, 8 elements/thread (b128 in, b128 out)
// ---------------------------------------------------------------------------
__global__ __launch_bounds__(256)
void convert_x_f16(const float* __restrict__ x, _Float16* __restrict__ x16) {
  const size_t i = ((size_t)blockIdx.x * blockDim.x + threadIdx.x) * 8;
  const v4f a = *(const v4f*)&x[i];
  const v4f b = *(const v4f*)&x[i + 4];
  v8h h;
#pragma unroll
  for (int e = 0; e < 4; ++e) { h[e] = (_Float16)a[e]; h[e + 4] = (_Float16)b[e]; }
  *(v8h*)&x16[i] = h;
}

// ---------------------------------------------------------------------------
// Pre-pass 2: GPTQ dequant W -> f16 [N][K] (k-contiguous per n)
// one int4 (4 packed bytes = 8 weights) per thread
// ---------------------------------------------------------------------------
__global__ __launch_bounds__(256)
void dequant_w_f16(const int* __restrict__ pw, const float* __restrict__ scales,
                   const float* __restrict__ zeros, _Float16* __restrict__ w16,
                   int K, int G) {
  const size_t i  = (size_t)blockIdx.x * blockDim.x + threadIdx.x;
  const int    K8 = K >> 3;                 // int4-chunks per row
  const size_t n  = i / K8;
  const int    e  = (int)(i % K8);          // chunk within row -> k0 = e*8
  const int    g  = e >> 3;                 // (e*8)/64
  const v4i   b4  = *(const v4i*)&pw[n * (size_t)(K >> 1) + (size_t)e * 4];
  const float s   = scales[n * G + g];
  const float zs  = zeros [n * G + g] * s;  // (q-z)*s = q*s - z*s
  v8h h;
#pragma unroll
  for (int t = 0; t < 4; ++t) {
    const int b = b4[t];
    h[2 * t]     = (_Float16)((float)(b        & 15) * s - zs);
    h[2 * t + 1] = (_Float16)((float)((b >> 4) & 15) * s - zs);
  }
  *(v8h*)&w16[n * (size_t)K + (size_t)e * 8] = h;
}

// ---------------------------------------------------------------------------
// Main GEMM: pure f16 WMMA, A16 [M][K] x B16 [N][K] -> out [M][N] + bias
// 128x128 block, 8 waves of 64x32, K-step 32, double-buffered LDS.
// Tile movement uses GLOBAL_LOAD_ASYNC_TO_LDS_B128 (ASYNCcnt) when available.
// ---------------------------------------------------------------------------
__global__ __launch_bounds__(256)
void gemm_f16_wmma(const _Float16* __restrict__ A16,
                   const _Float16* __restrict__ B16,
                   const float*    __restrict__ bias,
                   float*          __restrict__ out,
                   int M, int N, int K) {
  __shared__ _Float16 As[2][TILE_M][LDS_STR];
  __shared__ _Float16 Bs[2][TILE_N][LDS_STR];

  const int tid  = threadIdx.x;
  const int lane = tid & 31;
  const int wid  = tid >> 5;
  const int lh   = lane & 15;
  const int ksel = lane >> 4;            // 0: lanes 0-15, 1: lanes 16-31

  const int mBlk = blockIdx.y * TILE_M;
  const int nBlk = blockIdx.x * TILE_N;

  // Tile loads: 4 threads per 32-half row (8 halves = 16B each), 64 rows/pass, 2 passes.
  const int tRow = tid >> 2;
  const int tCol = (tid & 3) * 8;

  const int KT = K / TILE_K;

#if USE_ASYNC_LDS
  auto issue_async = [&](int buf, int k0) {
#pragma unroll
    for (int i = 0; i < 2; ++i) {
      __builtin_amdgcn_global_load_async_to_lds_b128(
          (as1_v4i_ptr)(as1_void_ptr)&A16[(size_t)(mBlk + tRow + 64 * i) * K + k0 + tCol],
          (as3_v4i_ptr)(as3_void_ptr)&As[buf][tRow + 64 * i][tCol],
          0, 0);
      __builtin_amdgcn_global_load_async_to_lds_b128(
          (as1_v4i_ptr)(as1_void_ptr)&B16[(size_t)(nBlk + tRow + 64 * i) * K + k0 + tCol],
          (as3_v4i_ptr)(as3_void_ptr)&Bs[buf][tRow + 64 * i][tCol],
          0, 0);
    }
  };
#else
  v8h aReg[2], bReg[2];
  auto fetch = [&](int k0) {
#pragma unroll
    for (int i = 0; i < 2; ++i) {
      aReg[i] = *(const v8h*)&A16[(size_t)(mBlk + tRow + 64 * i) * K + k0 + tCol];
      bReg[i] = *(const v8h*)&B16[(size_t)(nBlk + tRow + 64 * i) * K + k0 + tCol];
    }
  };
  auto stage = [&](int buf) {
#pragma unroll
    for (int i = 0; i < 2; ++i) {
      *(v8h*)&As[buf][tRow + 64 * i][tCol] = aReg[i];
      *(v8h*)&Bs[buf][tRow + 64 * i][tCol] = bReg[i];
    }
  };
#endif

  const int mBase = (wid & 1) * 64;
  const int nBase = (wid >> 1) * 32;

  v8f acc[4][2];
  const v8f vzero = {};
#pragma unroll
  for (int mi = 0; mi < 4; ++mi)
#pragma unroll
    for (int ni = 0; ni < 2; ++ni) acc[mi][ni] = vzero;

#if USE_ASYNC_LDS
  issue_async(0, 0);
  __builtin_amdgcn_s_wait_asynccnt(0);
  __syncthreads();
#else
  fetch(0);
  stage(0);
  __syncthreads();
#endif

  for (int kt = 0; kt < KT; ++kt) {
    const int  cur  = kt & 1;
    const bool more = (kt + 1) < KT;

#if USE_ASYNC_LDS
    if (more) issue_async(cur ^ 1, (kt + 1) * TILE_K);
#else
    if (more) fetch((kt + 1) * TILE_K);
#endif
    if (kt + 2 < KT) {
      __builtin_prefetch(&A16[(size_t)(mBlk + tRow) * K + (kt + 2) * TILE_K + tCol], 0, 3);
      __builtin_prefetch(&B16[(size_t)(nBlk + tRow) * K + (kt + 2) * TILE_K + tCol], 0, 3);
    }

    // A fragments: 16x32 f16 layout (lanes 0-15: K 0-7 / 16-23; lanes 16-31: K 8-15 / 24-31)
    v16h afr[4], bfr[2];
#pragma unroll
    for (int mi = 0; mi < 4; ++mi) {
      const int m = mBase + mi * 16 + lh;
      union { v16h v; v8h h[2]; } u;
      u.h[0] = *(const v8h*)&As[cur][m][ksel * 8];
      u.h[1] = *(const v8h*)&As[cur][m][ksel * 8 + 16];
      afr[mi] = u.v;
    }
    // B fragments: 32x16, lanes 0-15 hold K 0-7 / 16-23, lanes 16-31 hold K 8-15 / 24-31 (Bs n-major)
#pragma unroll
    for (int ni = 0; ni < 2; ++ni) {
      const int n = nBase + ni * 16 + lh;
      union { v16h v; v8h h[2]; } u;
      u.h[0] = *(const v8h*)&Bs[cur][n][ksel * 8];
      u.h[1] = *(const v8h*)&Bs[cur][n][ksel * 8 + 16];
      bfr[ni] = u.v;
    }

#pragma unroll
    for (int mi = 0; mi < 4; ++mi)
#pragma unroll
      for (int ni = 0; ni < 2; ++ni)
        acc[mi][ni] = __builtin_amdgcn_wmma_f32_16x16x32_f16(
            false, afr[mi], false, bfr[ni], (short)0, acc[mi][ni], false, false);

#if USE_ASYNC_LDS
    if (more) __builtin_amdgcn_s_wait_asynccnt(0);
    __syncthreads();
#else
    if (more) stage(cur ^ 1);
    __syncthreads();
#endif
  }

  // Epilogue: C 16x16 f32 layout — VGPR r: lanes 0-15 -> M=r, lanes 16-31 -> M=r+8; N=lane&15.
  // Non-temporal stores: keep the L2-resident x16/w16 working set from being evicted.
#pragma unroll
  for (int ni = 0; ni < 2; ++ni) {
    const int   col = nBlk + nBase + ni * 16 + lh;
    const float bv  = bias[col];
#pragma unroll
    for (int mi = 0; mi < 4; ++mi) {
#pragma unroll
      for (int r = 0; r < 8; ++r) {
        const int row = mBlk + mBase + mi * 16 + r + ksel * 8;
        __builtin_nontemporal_store(acc[mi][ni][r] + bv, &out[(size_t)row * N + col]);
      }
    }
  }
}

// ---------------------------------------------------------------------------
// Fallback: fully fused (dequant inside the GEMM loop) for small workspaces
// ---------------------------------------------------------------------------
__global__ __launch_bounds__(256)
void gptq_wmma_fused(const float* __restrict__ x,
                     const int*   __restrict__ pw,
                     const float* __restrict__ scales,
                     const float* __restrict__ zeros,
                     const float* __restrict__ bias,
                     float*       __restrict__ out,
                     int M, int N, int K, int G) {
  __shared__ _Float16 As[2][TILE_M][LDS_STR];
  __shared__ _Float16 Bs[2][TILE_N][LDS_STR];

  const int tid  = threadIdx.x;
  const int lane = tid & 31;
  const int wid  = tid >> 5;
  const int lh   = lane & 15;
  const int ksel = lane >> 4;

  const int mBlk = blockIdx.y * TILE_M;
  const int nBlk = blockIdx.x * TILE_N;

  const int aRow = tid >> 3;
  const int aCol = (tid & 7) * 4;
  const int bRow = tid >> 2;
  const int bCol = (tid & 3) * 4;

  const int KT    = K / TILE_K;
  const int halfK = K >> 1;

  v4f   aReg[4];
  v4i   bReg[2];
  float sReg[2], zReg[2];

  auto global_fetch = [&](int k0) {
#pragma unroll
    for (int i = 0; i < 4; ++i)
      aReg[i] = *(const v4f*)&x[(size_t)(mBlk + aRow + 32 * i) * K + k0 + aCol];
    const int g = k0 >> 6;
#pragma unroll
    for (int i = 0; i < 2; ++i) {
      const int n = nBlk + bRow + 64 * i;
      bReg[i] = *(const v4i*)&pw[(size_t)n * halfK + (k0 >> 1) + bCol];
      sReg[i] = scales[(size_t)n * G + g];
      zReg[i] = zeros [(size_t)n * G + g];
    }
  };

  auto lds_store = [&](int buf) {
#pragma unroll
    for (int i = 0; i < 4; ++i) {
      v4h h;
#pragma unroll
      for (int e = 0; e < 4; ++e) h[e] = (_Float16)aReg[i][e];
      *(v4h*)&As[buf][aRow + 32 * i][aCol] = h;
    }
#pragma unroll
    for (int i = 0; i < 2; ++i) {
      const float s  = sReg[i];
      const float zs = zReg[i] * s;
#pragma unroll
      for (int t = 0; t < 4; ++t) {
        const int b = bReg[i][t];
        v2h p;
        p[0] = (_Float16)((float)(b        & 15) * s - zs);
        p[1] = (_Float16)((float)((b >> 4) & 15) * s - zs);
        *(v2h*)&Bs[buf][bRow + 64 * i][(bCol + t) * 2] = p;
      }
    }
  };

  const int mBase = (wid & 1) * 64;
  const int nBase = (wid >> 1) * 32;

  v8f acc[4][2];
  const v8f vzero = {};
#pragma unroll
  for (int mi = 0; mi < 4; ++mi)
#pragma unroll
    for (int ni = 0; ni < 2; ++ni) acc[mi][ni] = vzero;

  global_fetch(0);
  lds_store(0);
  __syncthreads();

  for (int kt = 0; kt < KT; ++kt) {
    const int  cur  = kt & 1;
    const bool more = (kt + 1) < KT;

    if (more) global_fetch((kt + 1) * TILE_K);

    v16h afr[4], bfr[2];
#pragma unroll
    for (int mi = 0; mi < 4; ++mi) {
      const int m = mBase + mi * 16 + lh;
      union { v16h v; v8h h[2]; } u;
      u.h[0] = *(const v8h*)&As[cur][m][ksel * 8];
      u.h[1] = *(const v8h*)&As[cur][m][ksel * 8 + 16];
      afr[mi] = u.v;
    }
#pragma unroll
    for (int ni = 0; ni < 2; ++ni) {
      const int n = nBase + ni * 16 + lh;
      union { v16h v; v8h h[2]; } u;
      u.h[0] = *(const v8h*)&Bs[cur][n][ksel * 8];
      u.h[1] = *(const v8h*)&Bs[cur][n][ksel * 8 + 16];
      bfr[ni] = u.v;
    }

#pragma unroll
    for (int mi = 0; mi < 4; ++mi)
#pragma unroll
      for (int ni = 0; ni < 2; ++ni)
        acc[mi][ni] = __builtin_amdgcn_wmma_f32_16x16x32_f16(
            false, afr[mi], false, bfr[ni], (short)0, acc[mi][ni], false, false);

    if (more) lds_store(cur ^ 1);
    __syncthreads();
  }

#pragma unroll
  for (int ni = 0; ni < 2; ++ni) {
    const int   col = nBlk + nBase + ni * 16 + lh;
    const float bv  = bias[col];
#pragma unroll
    for (int mi = 0; mi < 4; ++mi) {
#pragma unroll
      for (int r = 0; r < 8; ++r) {
        const int row = mBlk + mBase + mi * 16 + r + ksel * 8;
        out[(size_t)row * N + col] = acc[mi][ni][r] + bv;
      }
    }
  }
}

// ---------------------------------------------------------------------------
extern "C" void kernel_launch(void* const* d_in, const int* in_sizes, int n_in,
                              void* d_out, int out_size, void* d_ws, size_t ws_size,
                              hipStream_t stream) {
  const float* x      = (const float*)d_in[0];
  const int*   pw     = (const int*)  d_in[1];
  const float* scales = (const float*)d_in[2];
  const float* zeros  = (const float*)d_in[3];
  const float* bias   = (const float*)d_in[4];
  float*       out    = (float*)d_out;

  const int N = in_sizes[4];
  const int G = in_sizes[2] / N;
  const int K = G * 64;
  const int M = (int)((long long)in_sizes[0] / K);

  dim3 grid(N / TILE_N, M / TILE_M);

  const size_t needWs = ((size_t)M * K + (size_t)N * K) * sizeof(_Float16);
  if (ws_size >= needWs) {
    _Float16* x16 = (_Float16*)d_ws;
    _Float16* w16 = (_Float16*)((char*)d_ws + (size_t)M * K * sizeof(_Float16));

    const size_t xThreads = (size_t)M * K / 8;
    const size_t wThreads = (size_t)N * K / 8;
    convert_x_f16<<<dim3((unsigned)(xThreads / 256)), 256, 0, stream>>>(x, x16);
    dequant_w_f16<<<dim3((unsigned)(wThreads / 256)), 256, 0, stream>>>(pw, scales, zeros, w16, K, G);
    gemm_f16_wmma<<<grid, 256, 0, stream>>>(x16, w16, bias, out, M, N, K);
  } else {
    gptq_wmma_fused<<<grid, 256, 0, stream>>>(x, pw, scales, zeros, bias, out, M, N, K, G);
  }
}
